// MoE_29291676959130
// MI455X (gfx1250) — compile-verified, matrix operand order checked by reading, not run on hardware
//
#include <hip/hip_runtime.h>
#include <hip/hip_bf16.h>

#define B_TOK 4096
#define DDIM  2048
#define HDIM  4096
#define NEXP  8
#define GATE_IN (2 * DDIM)

typedef __attribute__((ext_vector_type(16))) __bf16 v16bf;
typedef __attribute__((ext_vector_type(4)))  __bf16 v4bf;
typedef __attribute__((ext_vector_type(4)))  float  v4f;
typedef __attribute__((ext_vector_type(8)))  float  v8f;

union FragBF {
    unsigned int u[8];
    v16bf v;
};

// float4 -> 4 packed bf16 (fptrunc <4 x float> -> <4 x bfloat>: backend
// selects paired v_cvt_pk_bf16_f32 instead of scalar cvt + repack)
__device__ __forceinline__ uint2 pack4_bf16(float4 f) {
    v4f vf = {f.x, f.y, f.z, f.w};
    v4bf vb = __builtin_convertvector(vf, v4bf);
    uint2 r;
    __builtin_memcpy(&r, &vb, 8);
    return r;
}

// ---------------------------------------------------------------------------
// Kernel 1: zero the output accumulator and the per-expert counters.
// ---------------------------------------------------------------------------
__global__ void moe_init_kernel(float* __restrict__ out, int* __restrict__ counts) {
    size_t idx = (size_t)blockIdx.x * blockDim.x + threadIdx.x;
    size_t n4 = (size_t)B_TOK * HDIM / 4;
    if (idx < n4) {
        ((float4*)out)[idx] = make_float4(0.f, 0.f, 0.f, 0.f);
    }
    if (blockIdx.x == 0 && threadIdx.x < NEXP) counts[threadIdx.x] = 0;
}

// ---------------------------------------------------------------------------
// Kernel 2: gating. One wave32 per token; float4-vectorized dot products over
// concat(x, emb[label]); lane reduction; top-2 + softmax; per-expert gather
// list append via atomics.
// ---------------------------------------------------------------------------
__global__ void __launch_bounds__(256) moe_gate_kernel(
    const float* __restrict__ x, const int* __restrict__ lbl,
    const float* __restrict__ emb, const float* __restrict__ gW,
    const float* __restrict__ gb, int* __restrict__ topk_out,
    int* __restrict__ counts, int* __restrict__ tokList,
    float* __restrict__ probList)
{
    int wave = threadIdx.x >> 5;
    int lane = threadIdx.x & 31;
    int t = blockIdx.x * 8 + wave;
    if (t >= B_TOK) return;

    const float* xr = x + (size_t)t * DDIM;
    const float* er = emb + (size_t)lbl[t] * DDIM;

    float acc[NEXP];
#pragma unroll
    for (int e = 0; e < NEXP; ++e) acc[e] = 0.f;

    // each lane owns 4 consecutive floats; 128-float stride per iteration.
    // DDIM (2048) is a multiple of 128, so the x/emb split is chunk-uniform.
    for (int c0 = lane * 4; c0 < GATE_IN; c0 += 128) {
        float4 v = (c0 < DDIM) ? *(const float4*)(xr + c0)
                               : *(const float4*)(er + (c0 - DDIM));
#pragma unroll
        for (int e = 0; e < NEXP; ++e) {
            float4 w = *(const float4*)(gW + (size_t)e * GATE_IN + c0);
            acc[e] += v.x * w.x + v.y * w.y + v.z * w.z + v.w * w.w;
        }
    }
#pragma unroll
    for (int e = 0; e < NEXP; ++e) {
#pragma unroll
        for (int off = 16; off >= 1; off >>= 1)
            acc[e] += __shfl_xor(acc[e], off, 32);
    }

    if (lane == 0) {
        float logit[NEXP];
#pragma unroll
        for (int e = 0; e < NEXP; ++e) logit[e] = acc[e] + gb[e];
        int i0 = 0; float w0 = logit[0];
#pragma unroll
        for (int e = 1; e < NEXP; ++e)
            if (logit[e] > w0) { w0 = logit[e]; i0 = e; }
        int i1 = (i0 == 0) ? 1 : 0; float w1 = logit[i1];
#pragma unroll
        for (int e = 0; e < NEXP; ++e) {
            if (e == i0) continue;
            if (logit[e] > w1) { w1 = logit[e]; i1 = e; }
        }
        float p0 = 1.f / (1.f + __expf(w1 - w0));
        float p1 = 1.f - p0;

        topk_out[t * 2 + 0] = i0;
        topk_out[t * 2 + 1] = i1;

        int s0 = atomicAdd(&counts[i0], 1);
        tokList[i0 * B_TOK + s0] = t;
        probList[i0 * B_TOK + s0] = p0;
        int s1 = atomicAdd(&counts[i1], 1);
        tokList[i1 * B_TOK + s1] = t;
        probList[i1 * B_TOK + s1] = p1;
    }
}

// ---------------------------------------------------------------------------
// Kernel 3: per-expert gathered GEMM.  BM=128 x BN=64 x BK=64; 8 waves (4x2);
// 2x2 16x16 f32 accumulators per wave; 8 v_wmma_f32_16x16x32_bf16 per staged
// tile.  Software-pipelined: double-buffered LDS, global loads for tile k+1
// issued before the WMMAs of tile k, one barrier per iteration; packed-cvt
// bf16 staging with ds_store_b128.
// ---------------------------------------------------------------------------
__global__ void __launch_bounds__(256, 2) moe_expert_gemm_kernel(
    const float* __restrict__ x, const float* __restrict__ eW,
    const float* __restrict__ eb, const int* __restrict__ counts,
    const int* __restrict__ tokList, const float* __restrict__ probList,
    float* __restrict__ out)
{
    __shared__ __align__(16) unsigned int ldsA[2][128 * 32]; // 2 x (128 rows x 64 bf16)
    __shared__ __align__(16) unsigned int ldsB[2][64 * 32];  // 2 x ( 64 rows x 64 bf16)

    int e = blockIdx.z;
    int count = counts[e];
    int m0 = blockIdx.y * 128;
    if (m0 >= count) return;
    int n0 = blockIdx.x * 64;

    int tid = threadIdx.x;
    int lane = tid & 31;
    int wave = tid >> 5;       // 0..7
    int wave_m = wave >> 1;    // 0..3 -> 32-row band
    int wave_n = wave & 1;     // 0..1 -> 32-col band
    int half = lane >> 4;
    int l16 = lane & 15;

    // A staging: 2 threads per row, 32 floats (8x float4) each
    int a_row = tid >> 1;
    int a_seg = tid & 1;
    int a_valid = (m0 + a_row) < count;
    int a_tok = a_valid ? tokList[e * B_TOK + m0 + a_row] : 0;
    const float* a_ptr = x + (size_t)a_tok * DDIM + a_seg * 32;

    // B staging: 4 threads per row, 16 floats (4x float4) each
    int b_row = tid >> 2;
    int b_seg = tid & 3;
    const float* b_ptr = eW + ((size_t)e * HDIM + (n0 + b_row)) * DDIM + b_seg * 16;

    v8f acc[2][2];
#pragma unroll
    for (int i = 0; i < 2; ++i)
#pragma unroll
        for (int j = 0; j < 2; ++j)
            acc[i][j] = (v8f){0.f, 0.f, 0.f, 0.f, 0.f, 0.f, 0.f, 0.f};

    // ---- prologue: fetch tile 0 (invalid rows stay zero forever) ----
    float4 fA[8], fB[4];
#pragma unroll
    for (int j = 0; j < 8; ++j) fA[j] = make_float4(0.f, 0.f, 0.f, 0.f);
    if (a_valid) {
        const float4* p = (const float4*)a_ptr;
#pragma unroll
        for (int j = 0; j < 8; ++j) fA[j] = p[j];
    }
    {
        const float4* q = (const float4*)b_ptr;
#pragma unroll
        for (int j = 0; j < 4; ++j) fB[j] = q[j];
    }

    const int nIter = DDIM / 64;
    for (int it = 0; it < nIter; ++it) {
        int buf = it & 1;

        // ---- pack current tile registers into LDS[buf] (b128 stores) ----
        {
            uint4* da = (uint4*)&ldsA[buf][a_row * 32 + a_seg * 16];
#pragma unroll
            for (int j = 0; j < 4; ++j) {
                uint2 lo = pack4_bf16(fA[2 * j]);
                uint2 hi = pack4_bf16(fA[2 * j + 1]);
                da[j] = make_uint4(lo.x, lo.y, hi.x, hi.y);
            }
            uint4* db = (uint4*)&ldsB[buf][b_row * 32 + b_seg * 8];
#pragma unroll
            for (int j = 0; j < 2; ++j) {
                uint2 lo = pack4_bf16(fB[2 * j]);
                uint2 hi = pack4_bf16(fB[2 * j + 1]);
                db[j] = make_uint4(lo.x, lo.y, hi.x, hi.y);
            }
        }
        __syncthreads();

        // ---- issue global loads for next tile; wait lands next iteration ----
        if (it + 1 < nIter) {
            int k0 = (it + 1) * 64;
            if (a_valid) {
                const float4* p = (const float4*)(a_ptr + k0);
#pragma unroll
                for (int j = 0; j < 8; ++j) fA[j] = p[j];
            }
            const float4* q = (const float4*)(b_ptr + k0);
#pragma unroll
            for (int j = 0; j < 4; ++j) fB[j] = q[j];
        }

        // ---- two K=32 steps; 8 WMMAs per staged tile ----
#pragma unroll
        for (int ks = 0; ks < 2; ++ks) {
            FragBF afrag[2], bfrag[2];
#pragma unroll
            for (int tm = 0; tm < 2; ++tm) {
                int row = wave_m * 32 + tm * 16 + l16;
                const unsigned int* base = &ldsA[buf][row * 32 + ks * 16];
                ((uint4*)afrag[tm].u)[0] = *(const uint4*)(base + half * 4);
                ((uint4*)afrag[tm].u)[1] = *(const uint4*)(base + 8 + half * 4);
            }
#pragma unroll
            for (int tn = 0; tn < 2; ++tn) {
                int row = wave_n * 32 + tn * 16 + l16;
                const unsigned int* base = &ldsB[buf][row * 32 + ks * 16];
                ((uint4*)bfrag[tn].u)[0] = *(const uint4*)(base + half * 4);
                ((uint4*)bfrag[tn].u)[1] = *(const uint4*)(base + 8 + half * 4);
            }
#pragma unroll
            for (int tm = 0; tm < 2; ++tm)
#pragma unroll
                for (int tn = 0; tn < 2; ++tn)
                    acc[tm][tn] = __builtin_amdgcn_wmma_f32_16x16x32_bf16(
                        /*neg_a=*/false, afrag[tm].v,
                        /*neg_b=*/false, bfrag[tn].v,
                        /*c_mod=*/(short)0, acc[tm][tn],
                        /*reuse_a=*/false, /*reuse_b=*/false);
        }
    }

    // ---- epilogue: out[token, n] += p * (acc + bias) via atomic f32 ----
    float bias[2];
#pragma unroll
    for (int tn = 0; tn < 2; ++tn)
        bias[tn] = eb[(size_t)e * HDIM + n0 + wave_n * 32 + tn * 16 + l16];

#pragma unroll
    for (int tm = 0; tm < 2; ++tm) {
#pragma unroll
        for (int v = 0; v < 8; ++v) {
            int m_idx = m0 + wave_m * 32 + tm * 16 + v + 8 * half; // C layout: M = v + 8*(lane/16)
            if (m_idx < count) {
                int tok = tokList[e * B_TOK + m_idx];
                float p = probList[e * B_TOK + m_idx];
#pragma unroll
                for (int tn = 0; tn < 2; ++tn) {
                    int n = n0 + wave_n * 32 + tn * 16 + l16;       // C layout: N = lane%16
                    float val = acc[tm][tn][v] + bias[tn];
                    atomicAdd(&out[(size_t)tok * HDIM + n], p * val);
                }
            }
        }
    }
}

// ---------------------------------------------------------------------------
extern "C" void kernel_launch(void* const* d_in, const int* in_sizes, int n_in,
                              void* d_out, int out_size, void* d_ws, size_t ws_size,
                              hipStream_t stream) {
    const float* x   = (const float*)d_in[0];
    const int*   lbl = (const int*)d_in[1];
    const float* emb = (const float*)d_in[2];
    const float* gW  = (const float*)d_in[3];
    const float* gb  = (const float*)d_in[4];
    const float* eW  = (const float*)d_in[5];
    const float* eb  = (const float*)d_in[6];

    float* out  = (float*)d_out;
    int*   topk = (int*)(out + (size_t)B_TOK * HDIM);   // second tuple output

    // workspace layout (~262 KB)
    int*   counts   = (int*)d_ws;
    int*   tokList  = (int*)((char*)d_ws + 256);
    float* probList = (float*)((char*)d_ws + 256 + sizeof(int) * NEXP * B_TOK);

    // 1) zero out + counters
    size_t n4 = (size_t)B_TOK * HDIM / 4;
    moe_init_kernel<<<(unsigned)((n4 + 255) / 256), 256, 0, stream>>>(out, counts);

    // 2) gating + top-2 + token gather
    moe_gate_kernel<<<B_TOK / 8, 256, 0, stream>>>(x, lbl, emb, gW, gb,
                                                   topk, counts, tokList, probList);

    // 3) per-expert gathered GEMM (expert slowest -> per-expert W stays in L2)
    dim3 grid(HDIM / 64, B_TOK / 128, NEXP);
    moe_expert_gemm_kernel<<<grid, 256, 0, stream>>>(x, eW, eb, counts,
                                                     tokList, probList, out);

    (void)in_sizes; (void)n_in; (void)out_size; (void)ws_size;
}
